// NERNet_45183055954057
// MI455X (gfx1250) — matrix operand order repaired and checked by reading, not compiled, not run on hardware
//
#include <hip/hip_runtime.h>
#include <hip/hip_bf16.h>

#define DEV static __device__ __forceinline__

typedef __attribute__((ext_vector_type(16))) __bf16 v16bf;
typedef __attribute__((ext_vector_type(8)))  float   v8f;

// ---------------------------------------------------------------- constants
#define NB   64        // batch
#define NT   512       // seq len
#define NE   256       // embed concat (2*128)
#define NH   384       // hidden
#define NG   1536      // 4*H
#define NK   13        // tags
#define BT   (NB*NT)   // 32768 flat rows

// ---------------------------------------------------------------- WMMA core
DEV v8f wmma_bf16(v16bf a, v16bf b, v8f c) {
  // D = A(16x32 bf16) * B(32x16 bf16) + C(16x16 f32)
  return __builtin_amdgcn_wmma_f32_16x16x32_bf16(false, a, false, b,
                                                 (short)0, c, false, false);
}

// A tile (16x32, MxK) from row-major bf16 [ld = row stride in elements].
// Layout used here (contiguous k, not the ISA table order): lanes 0-15 -> M=lane, VGPR v: K = (lane/16)*16 + v*2
DEV v16bf load_a_tile(const __bf16* __restrict__ src, int ld, int row0, int k0, int lane) {
  int m = lane & 15, half = lane >> 4;
  const __bf16* r = src + (size_t)(row0 + m) * ld + k0;
  v16bf a;
#pragma unroll
  for (int v = 0; v < 8; ++v) {
    int k = half * 16 + (v << 1);
    a[2 * v]     = r[k];
    a[2 * v + 1] = r[k + 1];
  }
  return a;
}

// B tile (32x16, KxN) where memory holds W row-major as (N x K): B(k,n) = W[n,k].
// Layout used here (contiguous k, same as load_a_tile): N = lane%16, lanes 0-15 hold K=0..15, lanes 16-31 hold K=16..31,
// VGPR v: K = half*16 + 2v, 2v+1
DEV v16bf load_b_tile(const __bf16* __restrict__ w, int ld, int n0, int k0, int lane) {
  int n = lane & 15, half = lane >> 4;
  const __bf16* r = w + (size_t)(n0 + n) * ld + k0 + half * 16;
  v16bf b;
#pragma unroll
  for (int v = 0; v < 8; ++v) {
    b[2 * v]     = r[2 * v];
    b[2 * v + 1] = r[2 * v + 1];
  }
  return b;
}

// ---------------------------------------------------------------- prep kernels
__global__ void k_f32_to_bf16(const float* __restrict__ src, __bf16* __restrict__ dst, int n) {
  int i = blockIdx.x * blockDim.x + threadIdx.x;
  if (i < n) dst[i] = (__bf16)src[i];
}

// pad w_em (13x768) -> (16x768) bf16, rows 13..15 = 0
__global__ void k_wem_pad(const float* __restrict__ src, __bf16* __restrict__ dst) {
  int i = blockIdx.x * blockDim.x + threadIdx.x;
  if (i < 16 * 768) {
    int r = i / 768, c = i % 768;
    dst[i] = (__bf16)(r < NK ? src[r * 768 + c] : 0.f);
  }
}

// gather char/bichar embeddings -> x bf16 (BT x 256)
__global__ void k_gather(const int* __restrict__ char_id, const int* __restrict__ bichar_id,
                         const float* __restrict__ ctab, const float* __restrict__ btab,
                         __bf16* __restrict__ x) {
  int row = blockIdx.x;          // b*T + t
  int i = threadIdx.x;           // 0..255
  float v = (i < 128) ? ctab[(size_t)char_id[row] * 128 + i]
                      : btab[(size_t)bichar_id[row] * 128 + (i - 128)];
  x[(size_t)row * NE + i] = (__bf16)v;
}

// ---------------------------------------------------------------- input projection
// xg = x(BT x 256) @ W_ih^T(256 x 1536) + b
// block = 8 waves; M-tile staged in LDS; K-loop software-pipelined (double buffer).
__global__ __launch_bounds__(256)
void k_input_proj(const __bf16* __restrict__ x, const __bf16* __restrict__ w,
                  const float* __restrict__ bias, float* __restrict__ xg) {
  __shared__ __bf16 sA[16 * NE];                 // 8 KB, shared by all 8 waves
  int bid = blockIdx.x;
  int mt = bid / 3, ng = bid % 3;                // 2048 M-tiles x 3 N-groups(512)
  int m0 = mt * 16;

  const uint4* gsrc = (const uint4*)(x + (size_t)m0 * NE);
  uint4* ldst = (uint4*)sA;
  for (int i = threadIdx.x; i < (16 * NE * 2) / 16; i += 256) ldst[i] = gsrc[i];
  __syncthreads();

  int lane = threadIdx.x & 31, wv = threadIdx.x >> 5;
  int n0 = ng * 512 + wv * 64;                   // each wave: 16x64 output
  v8f acc[4] = {};

  v16bf a = load_a_tile(sA, NE, 0, 0, lane);
  v16bf b[4];
#pragma unroll
  for (int j = 0; j < 4; ++j) b[j] = load_b_tile(w, NE, n0 + j * 16, 0, lane);

#pragma unroll
  for (int ks = 0; ks < NE / 32; ++ks) {
    v16bf an, bn[4];
    if (ks < NE / 32 - 1) {                      // issue next-iter loads first
      an = load_a_tile(sA, NE, 0, (ks + 1) * 32, lane);
#pragma unroll
      for (int j = 0; j < 4; ++j) bn[j] = load_b_tile(w, NE, n0 + j * 16, (ks + 1) * 32, lane);
    }
#pragma unroll
    for (int j = 0; j < 4; ++j) acc[j] = wmma_bf16(a, b[j], acc[j]);
    if (ks < NE / 32 - 1) {
      a = an;
#pragma unroll
      for (int j = 0; j < 4; ++j) b[j] = bn[j];
    }
  }

  int mrow = (lane >> 4) << 3, ncol = lane & 15;
#pragma unroll
  for (int j = 0; j < 4; ++j) {
    float bv = bias[n0 + j * 16 + ncol];
#pragma unroll
    for (int v = 0; v < 8; ++v)
      xg[(size_t)(m0 + mrow + v) * NG + n0 + j * 16 + ncol] = acc[j][v] + bv;
  }
}

// ---------------------------------------------------------------- LSTM scan
// One workgroup (16 waves, 512 thr) per direction; blockIdx.x = dir.
// h lives in LDS (bf16, 48KB); c lives in registers (wave owns 6 fixed tiles all T).
// 16 waves (not 32) -> ~2x VGPR budget per wave for the pipelined tiles.
__global__ __launch_bounds__(512)
void k_lstm_scan(const float* __restrict__ xg_f, const float* __restrict__ xg_b,
                 const __bf16* __restrict__ whh_f, const __bf16* __restrict__ whh_b,
                 __bf16* __restrict__ hf, __bf16* __restrict__ hb) {
  __shared__ __bf16 hbuf[NB * NH];               // 48 KB
  int dir = blockIdx.x;
  const float*  xg  = dir ? xg_b  : xg_f;
  const __bf16* whh = dir ? whh_b : whh_f;
  __bf16*       ho  = dir ? hb    : hf;

  int lane = threadIdx.x & 31, wv = threadIdx.x >> 5;
  int mrow = (lane >> 4) << 3, ncol = lane & 15;

  int mt[6], jt[6];
#pragma unroll
  for (int i = 0; i < 6; ++i) { int g = wv * 6 + i; mt[i] = g / 24; jt[i] = g % 24; }

  float cst[6][8] = {};                          // cell state, persistent
  float hval[6][8];

  for (int t = 0; t < NT; ++t) {
    int tt = dir ? (NT - 1 - t) : t;

    // prefetch next timestep's xg stream (hide HBM behind recurrence GEMM)
    if (t + 1 < NT) {
      int tn = dir ? (NT - 2 - t) : (t + 1);
#pragma unroll
      for (int i = 0; i < 6; ++i)
        __builtin_prefetch(&xg[((size_t)(mt[i] * 16 + mrow) * NT + tn) * NG + jt[i] * 16 + ncol], 0, 1);
    }

#pragma unroll
    for (int i = 0; i < 6; ++i) {
      int m0 = mt[i] * 16, j0 = jt[i] * 16;
      v8f acc[4];
#pragma unroll
      for (int q = 0; q < 4; ++q) {              // init from xg_t (bias folded)
        int n = q * NH + j0 + ncol;
#pragma unroll
        for (int v = 0; v < 8; ++v)
          acc[q][v] = xg[((size_t)(m0 + mrow + v) * NT + tt) * NG + n];
      }
      if (t > 0) {                               // + h_{t-1} @ W_hh^T, pipelined
        v16bf a = load_a_tile(hbuf, NH, m0, 0, lane);
        v16bf b[4];
#pragma unroll
        for (int q = 0; q < 4; ++q) b[q] = load_b_tile(whh, NH, q * NH + j0, 0, lane);
#pragma unroll
        for (int ks = 0; ks < NH / 32; ++ks) {
          v16bf an, bn[4];
          if (ks < NH / 32 - 1) {
            an = load_a_tile(hbuf, NH, m0, (ks + 1) * 32, lane);
#pragma unroll
            for (int q = 0; q < 4; ++q)
              bn[q] = load_b_tile(whh, NH, q * NH + j0, (ks + 1) * 32, lane);
          }
#pragma unroll
          for (int q = 0; q < 4; ++q) acc[q] = wmma_bf16(a, b[q], acc[q]);
          if (ks < NH / 32 - 1) {
            a = an;
#pragma unroll
            for (int q = 0; q < 4; ++q) b[q] = bn[q];
          }
        }
      }
#pragma unroll
      for (int v = 0; v < 8; ++v) {
        float ig = 1.f / (1.f + __expf(-acc[0][v]));
        float fg = 1.f / (1.f + __expf(-acc[1][v]));
        float gg = tanhf(acc[2][v]);
        float og = 1.f / (1.f + __expf(-acc[3][v]));
        float c  = fg * cst[i][v] + ig * gg;
        cst[i][v]  = c;
        hval[i][v] = og * tanhf(c);
      }
    }
    __syncthreads();                             // all hbuf reads of step done
#pragma unroll
    for (int i = 0; i < 6; ++i) {
      int m0 = mt[i] * 16, j0 = jt[i] * 16;
#pragma unroll
      for (int v = 0; v < 8; ++v) {
        __bf16 hv = (__bf16)hval[i][v];
        int bb = m0 + mrow + v;
        hbuf[bb * NH + j0 + ncol] = hv;
        ho[((size_t)bb * NT + tt) * NH + j0 + ncol] = hv;
      }
    }
    __syncthreads();                             // new h visible for next step
  }
}

// ---------------------------------------------------------------- emission logits
// logits(BT x 16pad) = [h_f | h_b](BT x 768) @ w_em^T + b_em   (cols >=13 junk)
__global__ __launch_bounds__(256)
void k_emission(const __bf16* __restrict__ hf, const __bf16* __restrict__ hb,
                const __bf16* __restrict__ wem, const float* __restrict__ b_em,
                float* __restrict__ logits) {
  int tile = blockIdx.x * 8 + (threadIdx.x >> 5);
  int lane = threadIdx.x & 31;
  int row0 = tile * 16;
  v8f acc = {};

  v16bf a = load_a_tile(hf, NH, row0, 0, lane);
  v16bf b = load_b_tile(wem, 768, 0, 0, lane);
#pragma unroll
  for (int ks = 0; ks < 24; ++ks) {
    v16bf an, bn;
    if (ks < 23) {
      int kn = ks + 1;
      an = load_a_tile(kn < 12 ? hf : hb, NH, row0, (kn % 12) * 32, lane);
      bn = load_b_tile(wem, 768, 0, kn * 32, lane);
    }
    acc = wmma_bf16(a, b, acc);
    if (ks < 23) { a = an; b = bn; }
  }

  int mrow = (lane >> 4) << 3, ncol = lane & 15;
  float bv = (ncol < NK) ? b_em[ncol] : 0.f;
#pragma unroll
  for (int v = 0; v < 8; ++v)
    logits[(size_t)(row0 + mrow + v) * 16 + ncol] = acc[v] + bv;
}

// ---------------------------------------------------------------- CRF (one wave / batch row)
DEV float lse16(float x) {                       // logsumexp over 16-lane group
  float m = x;
#pragma unroll
  for (int s = 1; s < 16; s <<= 1) m = fmaxf(m, __shfl_xor(m, s, 32));
  float e = __expf(x - m);
#pragma unroll
  for (int s = 1; s < 16; s <<= 1) e += __shfl_xor(e, s, 32);
  return m + __logf(e);
}

__global__ __launch_bounds__(1024)
void k_crf(const float* __restrict__ logits, const int* __restrict__ char_id,
           const int* __restrict__ label_id, const float* __restrict__ start_trans,
           const float* __restrict__ end_trans, const float* __restrict__ trans,
           float* __restrict__ llh) {
  int wv = threadIdx.x >> 5, lane = threadIdx.x & 31;
  int b = blockIdx.x * 32 + wv;                  // 0..63
  int j = lane & 15;                             // lanes 16-31 mirror 0-15

  float tcol[NK];
#pragma unroll
  for (int i = 0; i < NK; ++i) tcol[i] = (j < NK) ? trans[i * NK + j] : -1e30f;

  const float* lg  = logits + (size_t)b * NT * 16;
  const int*   cid = char_id  + b * NT;
  const int*   tag = label_id + b * NT;

  // t = 0 : log-softmax + init
  float logit = (j < NK) ? lg[j] : -1e30f;
  float em    = logit - lse16(logit);
  float alpha = (j < NK) ? start_trans[j] + em : -1e30f;

  int   tg_prev = tag[0];
  float score   = start_trans[tg_prev] + __shfl(em, tg_prev, 32);
  int   maskcnt = (cid[0] != 0) ? 1 : 0;

  for (int t = 1; t < NT; ++t) {
    float lt = (j < NK) ? lg[t * 16 + j] : -1e30f;
    em = lt - lse16(lt);
    bool m_t = (cid[t] != 0);
    int  tg  = tag[t];
    if (m_t) score += trans[tg_prev * NK + tg] + __shfl(em, tg, 32);
    tg_prev = tg;
    maskcnt += m_t ? 1 : 0;
    // forward recursion: nxt_j = em_j + lse_i(alpha_i + trans[i,j])
    float mv = -1e30f, ss = 0.f;
#pragma unroll
    for (int i = 0; i < NK; ++i) {
      float v = __shfl(alpha, i, 32) + tcol[i];
      if (v > mv) { ss = ss * __expf(mv - v) + 1.f; mv = v; }
      else        { ss += __expf(v - mv); }
    }
    float nxt = mv + __logf(ss) + em;
    if (m_t && j < NK) alpha = nxt;
  }
  int se = maskcnt - 1; if (se < 0) se = 0;
  score += end_trans[tag[se]];
  float fin  = (j < NK) ? alpha + end_trans[j] : -1e30f;
  float logZ = lse16(fin);
  if (lane == 0) llh[b] = score - logZ;
}

__global__ void k_final(const float* __restrict__ llh, float* __restrict__ out) {
  int lane = threadIdx.x;                        // 32 threads
  float s = llh[lane] + llh[lane + 32];
#pragma unroll
  for (int off = 16; off; off >>= 1) s += __shfl_down(s, off, 32);
  if (lane == 0) out[0] = -s / (float)NB;
}

// ---------------------------------------------------------------- launch
extern "C" void kernel_launch(void* const* d_in, const int* in_sizes, int n_in,
                              void* d_out, int out_size, void* d_ws, size_t ws_size,
                              hipStream_t stream) {
  const int*   char_id   = (const int*)  d_in[0];
  const int*   bichar_id = (const int*)  d_in[1];
  const int*   label_id  = (const int*)  d_in[2];
  const float* char_tab  = (const float*)d_in[3];
  const float* bichar_tab= (const float*)d_in[4];
  const float* w_ih_f    = (const float*)d_in[5];
  const float* w_hh_f    = (const float*)d_in[6];
  const float* b_f       = (const float*)d_in[7];
  const float* w_ih_b    = (const float*)d_in[8];
  const float* w_hh_b    = (const float*)d_in[9];
  const float* b_b       = (const float*)d_in[10];
  const float* w_em      = (const float*)d_in[11];
  const float* b_em      = (const float*)d_in[12];
  const float* start_tr  = (const float*)d_in[13];
  const float* end_tr    = (const float*)d_in[14];
  const float* trans     = (const float*)d_in[15];
  float* out = (float*)d_out;

  char* base = (char*)d_ws;
  size_t off = 0;
  auto take = [&](size_t bytes) -> void* {
    void* p = base + off;
    off += (bytes + 255) & ~(size_t)255;
    return p;
  };
  float*  xg_f   = (float*) take((size_t)BT * NG * 4);
  float*  xg_b   = (float*) take((size_t)BT * NG * 4);
  __bf16* x_bf   = (__bf16*)take((size_t)BT * NE * 2);
  __bf16* wihf   = (__bf16*)take((size_t)NG * NE * 2);
  __bf16* wihb   = (__bf16*)take((size_t)NG * NE * 2);
  __bf16* whhf   = (__bf16*)take((size_t)NG * NH * 2);
  __bf16* whhb   = (__bf16*)take((size_t)NG * NH * 2);
  __bf16* wemb   = (__bf16*)take((size_t)16 * 768 * 2);
  __bf16* hf     = (__bf16*)take((size_t)BT * NH * 2);
  __bf16* hb     = (__bf16*)take((size_t)BT * NH * 2);
  float*  logits = (float*) take((size_t)BT * 16 * 4);
  float*  llh    = (float*) take((size_t)NB * 4);

  // 1. weight conversion to bf16
  int nih = NG * NE, nhh = NG * NH;
  k_f32_to_bf16<<<(nih + 255) / 256, 256, 0, stream>>>(w_ih_f, wihf, nih);
  k_f32_to_bf16<<<(nih + 255) / 256, 256, 0, stream>>>(w_ih_b, wihb, nih);
  k_f32_to_bf16<<<(nhh + 255) / 256, 256, 0, stream>>>(w_hh_f, whhf, nhh);
  k_f32_to_bf16<<<(nhh + 255) / 256, 256, 0, stream>>>(w_hh_b, whhb, nhh);
  k_wem_pad<<<(16 * 768 + 255) / 256, 256, 0, stream>>>(w_em, wemb);

  // 2. embedding gather
  k_gather<<<BT, 256, 0, stream>>>(char_id, bichar_id, char_tab, bichar_tab, x_bf);

  // 3. input projection GEMMs (WMMA, pipelined)
  k_input_proj<<<(BT / 16) * 3, 256, 0, stream>>>(x_bf, wihf, b_f, xg_f);
  k_input_proj<<<(BT / 16) * 3, 256, 0, stream>>>(x_bf, wihb, b_b, xg_b);

  // 4. bidirectional LSTM scan (WMMA recurrence, 1 WGP per direction)
  k_lstm_scan<<<2, 512, 0, stream>>>(xg_f, xg_b, whhf, whhb, hf, hb);

  // 5. emission logits (WMMA, pipelined)
  k_emission<<<BT / 16 / 8, 256, 0, stream>>>(hf, hb, wemb, b_em, logits);

  // 6. CRF NLL (one wave per batch row) + final reduce
  k_crf<<<2, 1024, 0, stream>>>(logits, char_id, label_id, start_tr, end_tr, trans, llh);
  k_final<<<1, 32, 0, stream>>>(llh, out);
}
